// MPNN_VN_32684701122621
// MI455X (gfx1250) — matrix operand order repaired and checked by reading, not compiled, not run on hardware
//
#include <hip/hip_runtime.h>
#include <hip/hip_bf16.h>
#include <math.h>

typedef _Float16 v16h __attribute__((ext_vector_type(16)));
typedef float    v8f  __attribute__((ext_vector_type(8)));

union fragu { v16h v; unsigned u[8]; };

__device__ __forceinline__ unsigned pack2(float lo, float hi) {
  // v_cvt_pk_rtz_f16_f32; result type is __fp16 ext_vector(2) -> bit-cast
  return __builtin_bit_cast(unsigned, __builtin_amdgcn_cvt_pkrtz(lo, hi));
}

// branch-free tanh: 1 - 2/(exp(2x)+1); exact saturation at +-1, TRANS-pipe exp
__device__ __forceinline__ float fast_tanh(float x) {
  float e = __builtin_amdgcn_exp2f(x * 2.885390081777927f);  // exp(2x)
  return 1.0f - 2.0f * __builtin_amdgcn_rcpf(e + 1.0f);
}

__device__ __forceinline__ v8f wmma16(v16h a, v16h b, v8f c) {
  return __builtin_amdgcn_wmma_f32_16x16x32_f16(false, a, false, b, (short)0, c,
                                                false, false);
}

__device__ __forceinline__ v8f splat8(float x) {
  v8f c;
#pragma unroll
  for (int g = 0; g < 8; ++g) c[g] = x;
  return c;
}

// ---- Fragment-major swizzle for packed-f16 B operands --------------------
// B 32x16 f16 frag: lane L = col n, kh = 8*(L>>4); VGPR i = K pair
// k = kb+kh+(i<4?2i:16+2(i-4)). Store so each lane's 8 pair-uints are contiguous:
// slot(k2, n) = ((j*ncols + n)*2 + ((r>>2)&1))*8 + (r&3) + 4*(r>>3), r = k2&15, j = k2>>4.
__device__ __forceinline__ int swzB(int k2, int n, int ncols) {
  int j = k2 >> 4, r = k2 & 15;
  return ((j * ncols + n) * 2 + ((r >> 2) & 1)) * 8 + (r & 3) + ((r >> 3) << 2);
}

// load B fragment (k-step j, absolute col n incl. lane&15): 2x ds_load_b128
__device__ __forceinline__ v16h load_fragB_swz(const unsigned* bp, int ncols,
                                               int j, int n) {
  int hb = (threadIdx.x & 31) >> 4;
  const uint4* p = (const uint4*)(bp + (size_t)((j * ncols + n) * 2 + hb) * 8);
  uint4 a = p[0], b = p[1];
  fragu r;
  r.u[0] = a.x; r.u[1] = a.y; r.u[2] = a.z; r.u[3] = a.w;
  r.u[4] = b.x; r.u[5] = b.y; r.u[6] = b.z; r.u[7] = b.w;
  return r.v;
}

// ---- A fragment loaders (documented 16-bit A 16x32 layout) ---------------
// lane L: row M=L&15; kh=8*(L>>4); VGPR i pairs: k = k0+kh+(i<4?2i:16+2(i-4))
__device__ __forceinline__ v16h load_fragA_full(const float* rowp, int k0,
                                                float scale) {
  int kh = ((threadIdx.x & 31) >> 4) * 8;
  const float4* p = (const float4*)(rowp + k0 + kh);
  const float4* q = (const float4*)(rowp + k0 + kh + 16);
  float4 x0 = p[0], x1 = p[1], y0 = q[0], y1 = q[1];
  fragu r;
  r.u[0] = pack2(x0.x * scale, x0.y * scale);
  r.u[1] = pack2(x0.z * scale, x0.w * scale);
  r.u[2] = pack2(x1.x * scale, x1.y * scale);
  r.u[3] = pack2(x1.z * scale, x1.w * scale);
  r.u[4] = pack2(y0.x * scale, y0.y * scale);
  r.u[5] = pack2(y0.z * scale, y0.w * scale);
  r.u[6] = pack2(y1.x * scale, y1.y * scale);
  r.u[7] = pack2(y1.z * scale, y1.w * scale);
  return r.v;
}

// K=16 (zero-padded to 32): every lane loads 8 consecutive floats at kh
__device__ __forceinline__ v16h load_fragA_k16(const float* rowp) {
  int kh = ((threadIdx.x & 31) >> 4) * 8;
  const float4* p = (const float4*)(rowp + kh);
  float4 x0 = p[0], x1 = p[1];
  fragu r;
  r.u[0] = pack2(x0.x, x0.y); r.u[1] = pack2(x0.z, x0.w);
  r.u[2] = pack2(x1.x, x1.y); r.u[3] = pack2(x1.z, x1.w);
  r.u[4] = 0u; r.u[5] = 0u; r.u[6] = 0u; r.u[7] = 0u;
  return r.v;
}

// K=8 (zero-padded): only lanes 0-15 carry data
__device__ __forceinline__ v16h load_fragA_k8(const float* rowp) {
  int lane = threadIdx.x & 31;
  fragu r;
#pragma unroll
  for (int i = 0; i < 8; ++i) r.u[i] = 0u;
  if (lane < 16) {
    const float4* p = (const float4*)rowp;
    float4 x0 = p[0], x1 = p[1];
    r.u[0] = pack2(x0.x, x0.y); r.u[1] = pack2(x0.z, x0.w);
    r.u[2] = pack2(x1.x, x1.y); r.u[3] = pack2(x1.z, x1.w);
  }
  return r.v;
}

// A frag from per-wave LDS staging (16 rows x 16 pair-uints, row-major)
__device__ __forceinline__ v16h load_fragA_stage(const unsigned* stu) {
  int lane = threadIdx.x & 31;
  int m = lane & 15;
  int kh2 = (lane >> 4) * 4;
  fragu r;
#pragma unroll
  for (int i = 0; i < 8; ++i) {
    int k2 = kh2 + (i < 4 ? i : 8 + (i - 4));
    r.u[i] = stu[m * 16 + k2];
  }
  return r.v;
}

// ==========================================================================
// Edge MLP (dominant): e_sum[e] = sum_n tanh( tanh(ef@W1+b1)@W2+b2 )
// One wave = 16 edges; L1: 16 WMMA (K=8 padded); L2: 32 WMMA (K=256, N=64).
// ==========================================================================
__global__ __launch_bounds__(256) void k_edge_mlp(
    const float* __restrict__ ef, const float* __restrict__ w1,
    const float* __restrict__ b1, const float* __restrict__ w2,
    const float* __restrict__ b2, float* __restrict__ e_sum) {
  __shared__ unsigned b1p[16 * 256];  // K padded to 32, fragment-major
  __shared__ float bias1[256];
  __shared__ unsigned b2p[128 * 64];  // 256x64, fragment-major
  __shared__ float bias2[64];
  __shared__ _Float16 stageh[8 * 16 * 32];

  for (int idx = threadIdx.x; idx < 16 * 256; idx += 256) {
    int k2 = idx >> 8, n = idx & 255;
    unsigned v = (k2 < 4) ? pack2(w1[(2 * k2) * 256 + n], w1[(2 * k2 + 1) * 256 + n])
                          : 0u;
    b1p[swzB(k2, n, 256)] = v;
  }
  for (int idx = threadIdx.x; idx < 128 * 64; idx += 256) {
    int k2 = idx >> 6, n = idx & 63;
    b2p[swzB(k2, n, 64)] = pack2(w2[(2 * k2) * 64 + n], w2[(2 * k2 + 1) * 64 + n]);
  }
  for (int i = threadIdx.x; i < 256; i += 256) bias1[i] = b1[i];
  if (threadIdx.x < 64) bias2[threadIdx.x] = b2[threadIdx.x];
  __syncthreads();

  int wave = threadIdx.x >> 5, lane = threadIdx.x & 31;
  int m = lane & 15, hb = lane >> 4;
  int e0 = (blockIdx.x * 8 + wave) * 16;

  v16h aedge = load_fragA_k8(ef + (size_t)(e0 + m) * 8);

  v8f acc2[4];
#pragma unroll
  for (int t2 = 0; t2 < 4; ++t2) acc2[t2] = splat8(bias2[t2 * 16 + m]);
  _Float16* sh = stageh + wave * (16 * 32);

  for (int j = 0; j < 8; ++j) {  // layer-2 K step = layer-1 N tiles 2j,2j+1
#pragma unroll
    for (int tt = 0; tt < 2; ++tt) {
      int n = (2 * j + tt) * 16 + m;
      v16h bf = load_fragB_swz(b1p, 256, 0, n);
      v8f c = wmma16(aedge, bf, splat8(bias1[n]));
#pragma unroll
      for (int g = 0; g < 8; ++g)
        sh[(g + 8 * hb) * 32 + tt * 16 + m] = (_Float16)fast_tanh(c[g]);
    }
    __threadfence_block();
    v16h a2 = load_fragA_stage((const unsigned*)sh);
#pragma unroll
    for (int t2 = 0; t2 < 4; ++t2) {
      v16h bf2 = load_fragB_swz(b2p, 64, j, t2 * 16 + m);
      acc2[t2] = wmma16(a2, bf2, acc2[t2]);
    }
    __threadfence_block();
  }

  float rs[8];
#pragma unroll
  for (int g = 0; g < 8; ++g) rs[g] = 0.f;
#pragma unroll
  for (int t2 = 0; t2 < 4; ++t2)
#pragma unroll
    for (int g = 0; g < 8; ++g) rs[g] += fast_tanh(acc2[t2][g]);
#pragma unroll
  for (int g = 0; g < 8; ++g) {
    float v = rs[g];
    v += __shfl_xor(v, 1, 32);
    v += __shfl_xor(v, 2, 32);
    v += __shfl_xor(v, 4, 32);
    v += __shfl_xor(v, 8, 32);
    if (m == g) e_sum[e0 + g + 8 * hb] = v;
  }
}

// ==========================================================================
// Node MLP: nh = tanh(tanh(nf@W1+b1)@W2+b2) -> hidden + alls[:,0:64]
// ==========================================================================
__global__ __launch_bounds__(256) void k_node_mlp(
    const float* __restrict__ nf, const float* __restrict__ w1,
    const float* __restrict__ b1, const float* __restrict__ w2,
    const float* __restrict__ b2, float* __restrict__ hidden,
    float* __restrict__ alls) {
  __shared__ unsigned b1p[16 * 256];
  __shared__ float bias1[256];
  __shared__ unsigned b2p[128 * 64];
  __shared__ float bias2[64];
  __shared__ _Float16 stageh[8 * 16 * 32];

  for (int idx = threadIdx.x; idx < 16 * 256; idx += 256) {
    int k2 = idx >> 8, n = idx & 255;
    unsigned v = (k2 < 8) ? pack2(w1[(2 * k2) * 256 + n], w1[(2 * k2 + 1) * 256 + n])
                          : 0u;
    b1p[swzB(k2, n, 256)] = v;
  }
  for (int idx = threadIdx.x; idx < 128 * 64; idx += 256) {
    int k2 = idx >> 6, n = idx & 63;
    b2p[swzB(k2, n, 64)] = pack2(w2[(2 * k2) * 64 + n], w2[(2 * k2 + 1) * 64 + n]);
  }
  for (int i = threadIdx.x; i < 256; i += 256) bias1[i] = b1[i];
  if (threadIdx.x < 64) bias2[threadIdx.x] = b2[threadIdx.x];
  __syncthreads();

  int wave = threadIdx.x >> 5, lane = threadIdx.x & 31;
  int m = lane & 15, hb = lane >> 4;
  int row0 = (blockIdx.x * 8 + wave) * 16;

  v16h afrag = load_fragA_k16(nf + (size_t)(row0 + m) * 16);

  v8f acc2[4];
#pragma unroll
  for (int t2 = 0; t2 < 4; ++t2) acc2[t2] = splat8(bias2[t2 * 16 + m]);
  _Float16* sh = stageh + wave * (16 * 32);

  for (int j = 0; j < 8; ++j) {
#pragma unroll
    for (int tt = 0; tt < 2; ++tt) {
      int n = (2 * j + tt) * 16 + m;
      v16h bf = load_fragB_swz(b1p, 256, 0, n);
      v8f c = wmma16(afrag, bf, splat8(bias1[n]));
#pragma unroll
      for (int g = 0; g < 8; ++g)
        sh[(g + 8 * hb) * 32 + tt * 16 + m] = (_Float16)fast_tanh(c[g]);
    }
    __threadfence_block();
    v16h a2 = load_fragA_stage((const unsigned*)sh);
#pragma unroll
    for (int t2 = 0; t2 < 4; ++t2) {
      v16h bf2 = load_fragB_swz(b2p, 64, j, t2 * 16 + m);
      acc2[t2] = wmma16(a2, bf2, acc2[t2]);
    }
    __threadfence_block();
  }
#pragma unroll
  for (int t2 = 0; t2 < 4; ++t2) {
    int n = t2 * 16 + m;
#pragma unroll
    for (int g = 0; g < 8; ++g) {
      int row = row0 + g + 8 * hb;
      float v = fast_tanh(acc2[t2][g]);
      hidden[row * 64 + n] = v;
      alls[row * 256 + n] = v;
    }
  }
}

// ==========================================================================
// [2048,64] @ [64,64] -> hg (no activation)
// ==========================================================================
__global__ __launch_bounds__(256) void k_gemm64(const float* __restrict__ X,
                                                const float* __restrict__ W,
                                                float* __restrict__ Y) {
  __shared__ unsigned wp[32 * 64];
  for (int idx = threadIdx.x; idx < 2048; idx += 256) {
    int k2 = idx >> 6, n = idx & 63;
    wp[swzB(k2, n, 64)] = pack2(W[(2 * k2) * 64 + n], W[(2 * k2 + 1) * 64 + n]);
  }
  __syncthreads();
  int wave = threadIdx.x >> 5, lane = threadIdx.x & 31;
  int m = lane & 15, hb = lane >> 4;
  int row0 = (blockIdx.x * 8 + wave) * 16;
  const float* rowp = X + (size_t)(row0 + m) * 64;
  v8f acc[4] = {splat8(0.f), splat8(0.f), splat8(0.f), splat8(0.f)};
  for (int ks = 0; ks < 2; ++ks) {
    v16h af = load_fragA_full(rowp, ks * 32, 1.0f);
#pragma unroll
    for (int t2 = 0; t2 < 4; ++t2) {
      v16h bf = load_fragB_swz(wp, 64, ks, t2 * 16 + m);
      acc[t2] = wmma16(af, bf, acc[t2]);
    }
  }
#pragma unroll
  for (int t2 = 0; t2 < 4; ++t2) {
    int n = t2 * 16 + m;
#pragma unroll
    for (int g = 0; g < 8; ++g) Y[(row0 + g + 8 * hb) * 64 + n] = acc[t2][g];
  }
}

// ==========================================================================
// out[b] = tanh( (rowscale? diag:I) * A[b](512x512) @ Bp[b](512x64 f16) )
// Bp stored chunk-major+fragment-major; LDS chunk = straight copy.
// ==========================================================================
__global__ __launch_bounds__(256) void k_spmm_tanh(
    const float* __restrict__ A, const float* __restrict__ rowscale,
    const unsigned* __restrict__ Bp, float* __restrict__ out) {
  __shared__ unsigned bch[64 * 64];
  int b = blockIdx.y;
  int wave = threadIdx.x >> 5, lane = threadIdx.x & 31;
  int m = lane & 15, hb = lane >> 4;
  int i0 = blockIdx.x * 128 + wave * 16;
  const float* rowp = A + ((size_t)b * 512 + i0 + m) * 512;
  float scale = rowscale ? rowscale[b * 512 + i0 + m] : 1.0f;
  const unsigned* Bpb = Bp + (size_t)b * 16384;
  v8f acc[4] = {splat8(0.f), splat8(0.f), splat8(0.f), splat8(0.f)};
  for (int kc = 0; kc < 4; ++kc) {
    for (int idx = threadIdx.x; idx < 4096; idx += 256)
      bch[idx] = Bpb[kc * 4096 + idx];
    __syncthreads();
    if (kc < 3) __builtin_prefetch(rowp + (kc + 1) * 128, 0, 1);
    for (int ks = 0; ks < 4; ++ks) {
      v16h af = load_fragA_full(rowp, kc * 128 + ks * 32, scale);
#pragma unroll
      for (int t2 = 0; t2 < 4; ++t2) {
        v16h bf = load_fragB_swz(bch, 64, ks, t2 * 16 + m);
        acc[t2] = wmma16(af, bf, acc[t2]);
      }
    }
    __syncthreads();
  }
#pragma unroll
  for (int t2 = 0; t2 < 4; ++t2) {
    int n = t2 * 16 + m;
#pragma unroll
    for (int g = 0; g < 8; ++g)
      out[((size_t)b * 512 + i0 + g + 8 * hb) * 64 + n] = fast_tanh(acc[t2][g]);
  }
}

// ==========================================================================
// comb: hidden = tanh([h2|gated](2048x128) @ W(128x64) + b) -> hidden, alls
// ==========================================================================
__global__ __launch_bounds__(256) void k_comb(
    const float* __restrict__ h2, const float* __restrict__ gated,
    const float* __restrict__ W, const float* __restrict__ bias,
    float* __restrict__ hidden, float* __restrict__ alls, int l) {
  __shared__ unsigned wp[64 * 64];
  __shared__ float bs[64];
  for (int idx = threadIdx.x; idx < 4096; idx += 256) {
    int k2 = idx >> 6, n = idx & 63;
    wp[swzB(k2, n, 64)] = pack2(W[(2 * k2) * 64 + n], W[(2 * k2 + 1) * 64 + n]);
  }
  if (threadIdx.x < 64) bs[threadIdx.x] = bias[threadIdx.x];
  __syncthreads();
  int wave = threadIdx.x >> 5, lane = threadIdx.x & 31;
  int m = lane & 15, hb = lane >> 4;
  int row0 = (blockIdx.x * 8 + wave) * 16;
  v8f acc[4];
#pragma unroll
  for (int t2 = 0; t2 < 4; ++t2) acc[t2] = splat8(bs[t2 * 16 + m]);
  for (int ks = 0; ks < 4; ++ks) {
    int k0 = ks * 32;
    const float* src = (k0 < 64) ? h2 + (size_t)(row0 + m) * 64 + k0
                                 : gated + (size_t)(row0 + m) * 64 + (k0 - 64);
    v16h af = load_fragA_full(src, 0, 1.0f);
#pragma unroll
    for (int t2 = 0; t2 < 4; ++t2) {
      v16h bf = load_fragB_swz(wp, 64, ks, t2 * 16 + m);
      acc[t2] = wmma16(af, bf, acc[t2]);
    }
  }
#pragma unroll
  for (int t2 = 0; t2 < 4; ++t2) {
    int n = t2 * 16 + m;
#pragma unroll
    for (int g = 0; g < 8; ++g) {
      int row = row0 + g + 8 * hb;
      float v = fast_tanh(acc[t2][g]);
      hidden[row * 64 + n] = v;
      alls[row * 256 + 64 * (l + 1) + n] = v;
    }
  }
}

// ==========================================================================
// lat1: lat_h = tanh(alls(2048x256) @ W(256x256) + b); grid.y = col quarter
// ==========================================================================
__global__ __launch_bounds__(256) void k_lat1(const float* __restrict__ alls,
                                              const float* __restrict__ W,
                                              const float* __restrict__ bias,
                                              float* __restrict__ lath) {
  __shared__ unsigned wp[128 * 64];
  __shared__ float bs[64];
  int q = blockIdx.y;
  for (int idx = threadIdx.x; idx < 8192; idx += 256) {
    int k2 = idx >> 6, nl = idx & 63, n = q * 64 + nl;
    wp[swzB(k2, nl, 64)] = pack2(W[(2 * k2) * 256 + n], W[(2 * k2 + 1) * 256 + n]);
  }
  if (threadIdx.x < 64) bs[threadIdx.x] = bias[q * 64 + threadIdx.x];
  __syncthreads();
  int wave = threadIdx.x >> 5, lane = threadIdx.x & 31;
  int m = lane & 15, hb = lane >> 4;
  int row0 = (blockIdx.x * 8 + wave) * 16;
  const float* rowp = alls + (size_t)(row0 + m) * 256;
  v8f acc[4];
#pragma unroll
  for (int t2 = 0; t2 < 4; ++t2) acc[t2] = splat8(bs[t2 * 16 + m]);
  for (int ks = 0; ks < 8; ++ks) {
    v16h af = load_fragA_full(rowp, ks * 32, 1.0f);
#pragma unroll
    for (int t2 = 0; t2 < 4; ++t2) {
      v16h bf = load_fragB_swz(wp, 64, ks, t2 * 16 + m);
      acc[t2] = wmma16(af, bf, acc[t2]);
    }
  }
#pragma unroll
  for (int t2 = 0; t2 < 4; ++t2) {
    int nl = t2 * 16 + m;
#pragma unroll
    for (int g = 0; g < 8; ++g)
      lath[(size_t)(row0 + g + 8 * hb) * 256 + q * 64 + nl] =
          fast_tanh(acc[t2][g]);
  }
}

// ==========================================================================
// lat2 fused node-sum: atomically accumulate sum_i tanh(lat_h@W2+b)
// ==========================================================================
__global__ __launch_bounds__(256) void k_lat2(const float* __restrict__ lath,
                                              const float* __restrict__ W,
                                              const float* __restrict__ bias,
                                              float* __restrict__ lat_sum) {
  __shared__ unsigned wp[128 * 64];
  __shared__ float bs[64];
  for (int idx = threadIdx.x; idx < 8192; idx += 256) {
    int k2 = idx >> 6, n = idx & 63;
    wp[swzB(k2, n, 64)] = pack2(W[(2 * k2) * 64 + n], W[(2 * k2 + 1) * 64 + n]);
  }
  if (threadIdx.x < 64) bs[threadIdx.x] = bias[threadIdx.x];
  __syncthreads();
  int wave = threadIdx.x >> 5, lane = threadIdx.x & 31;
  int m = lane & 15;
  int row0 = (blockIdx.x * 8 + wave) * 16;
  const float* rowp = lath + (size_t)(row0 + m) * 256;
  v8f acc[4];
#pragma unroll
  for (int t2 = 0; t2 < 4; ++t2) acc[t2] = splat8(bs[t2 * 16 + m]);
  for (int ks = 0; ks < 8; ++ks) {
    v16h af = load_fragA_full(rowp, ks * 32, 1.0f);
#pragma unroll
    for (int t2 = 0; t2 < 4; ++t2) {
      v16h bf = load_fragB_swz(wp, 64, ks, t2 * 16 + m);
      acc[t2] = wmma16(af, bf, acc[t2]);
    }
  }
  int bidx = row0 >> 9;
#pragma unroll
  for (int t2 = 0; t2 < 4; ++t2) {
    int n = t2 * 16 + m;
    float s = 0.f;
#pragma unroll
    for (int g = 0; g < 8; ++g) s += fast_tanh(acc[t2][g]);
    atomicAdd(&lat_sum[bidx * 64 + n], s);
  }
}

// ==========================================================================
// Small scalar kernels
// ==========================================================================
__global__ void k_dinv(const float* __restrict__ adj, float* __restrict__ dinv) {
  int r = blockIdx.x * blockDim.x + threadIdx.x;
  if (r >= 2048) return;
  const float4* rp = (const float4*)(adj + (size_t)r * 512);
  float s = 0.f;
  for (int j = 0; j < 128; ++j) {
    float4 x = rp[j];
    s += x.x + x.y + x.z + x.w;
  }
  dinv[r] = 1.0f / (s + 1.0f);
}

__global__ void k_zero(float* p, int n) {
  int i = blockIdx.x * blockDim.x + threadIdx.x;
  if (i < n) p[i] = 0.f;
}

// virtual-node MLP: mean-pool -> fc(64->128)+BN+relu -> fc(128->64)+BN+relu
__global__ __launch_bounds__(256) void k_vn(
    const float* __restrict__ h1, const float* __restrict__ w1,
    const float* __restrict__ b1, const float* __restrict__ g1,
    const float* __restrict__ be1, const float* __restrict__ w2,
    const float* __restrict__ b2, const float* __restrict__ g2,
    const float* __restrict__ be2, float* __restrict__ vn_out, int l) {
  __shared__ float p0[4 * 64], x1[4 * 128], x2[4 * 64];
  int t = threadIdx.x;
  {
    int b = t >> 6, c = t & 63;
    float s = 0.f;
    for (int i = 0; i < 512; ++i) s += h1[((size_t)b * 512 + i) * 64 + c];
    p0[t] = s * (1.0f / 512.0f);
  }
  __syncthreads();
  for (int idx = t; idx < 512; idx += 256) {
    int b = idx >> 7, c = idx & 127;
    const float* W = w1 + l * 64 * 128;
    float s = b1[l * 128 + c];
    for (int k = 0; k < 64; ++k) s += p0[b * 64 + k] * W[k * 128 + c];
    x1[idx] = s;
  }
  __syncthreads();
  if (t < 128) {
    float mn = 0.f;
    for (int b = 0; b < 4; ++b) mn += x1[b * 128 + t];
    mn *= 0.25f;
    float var = 0.f;
    for (int b = 0; b < 4; ++b) {
      float d = x1[b * 128 + t] - mn;
      var += d * d;
    }
    var *= 0.25f;
    float inv = rsqrtf(var + 1e-5f);
    float gg = g1[l * 128 + t], bb = be1[l * 128 + t];
    for (int b = 0; b < 4; ++b) {
      float y = gg * (x1[b * 128 + t] - mn) * inv + bb;
      x1[b * 128 + t] = y > 0.f ? y : 0.f;
    }
  }
  __syncthreads();
  {
    int b = t >> 6, c = t & 63;
    const float* W = w2 + l * 128 * 64;
    float s = b2[l * 64 + c];
    for (int k = 0; k < 128; ++k) s += x1[b * 128 + k] * W[k * 64 + c];
    x2[t] = s;
  }
  __syncthreads();
  if (t < 64) {
    float mn = 0.f;
    for (int b = 0; b < 4; ++b) mn += x2[b * 64 + t];
    mn *= 0.25f;
    float var = 0.f;
    for (int b = 0; b < 4; ++b) {
      float d = x2[b * 64 + t] - mn;
      var += d * d;
    }
    var *= 0.25f;
    float inv = rsqrtf(var + 1e-5f);
    float gg = g2[l * 64 + t], bb = be2[l * 64 + t];
    for (int b = 0; b < 4; ++b) {
      float y = gg * (x2[b * 64 + t] - mn) * inv + bb;
      vn_out[b * 64 + t] = y > 0.f ? y : 0.f;
    }
  }
}

// h2 = h1 + vn; emit packed f16 pairs in chunk-major/fragment-major layout
__global__ void k_addvn(const float* __restrict__ h1, const float* __restrict__ vn,
                        float* __restrict__ h2, unsigned* __restrict__ h2p) {
  int idx = blockIdx.x * blockDim.x + threadIdx.x;  // b*16384 + k2*64 + n
  if (idx >= 4 * 256 * 64) return;
  int n = idx & 63, k2 = (idx >> 6) & 255, b = idx >> 14;
  int r = b * 512 + 2 * k2;
  float vv = vn[b * 64 + n];
  float a0 = h1[(size_t)r * 64 + n] + vv;
  float a1 = h1[(size_t)(r + 1) * 64 + n] + vv;
  h2[(size_t)r * 64 + n] = a0;
  h2[(size_t)(r + 1) * 64 + n] = a1;
  h2p[b * 16384 + (k2 >> 6) * 4096 + swzB(k2 & 63, n, 64)] = pack2(a0, a1);
}

// pack [2048,64] f32 -> chunk-major/fragment-major f16 pairs
__global__ void k_pack(const float* __restrict__ X, unsigned* __restrict__ Xp) {
  int idx = blockIdx.x * blockDim.x + threadIdx.x;
  if (idx >= 65536) return;
  int n = idx & 63, p = idx >> 6;       // global pair row 0..1023
  int b = p >> 8, k2 = p & 255;         // batch, pair-row in batch
  unsigned v = pack2(X[(size_t)(2 * p) * 64 + n], X[(size_t)(2 * p + 1) * 64 + n]);
  Xp[b * 16384 + (k2 >> 6) * 4096 + swzB(k2 & 63, n, 64)] = v;
}

// final: h=tanh(lat_sum@fc1+b); predict=h@fc2+b; out=[predict(4)|lat_sum(256)]
__global__ __launch_bounds__(512) void k_fc(const float* __restrict__ lat_sum,
                                            const float* __restrict__ w1,
                                            const float* __restrict__ b1,
                                            const float* __restrict__ w2,
                                            const float* __restrict__ b2,
                                            float* __restrict__ out) {
  __shared__ float h[4 * 512];
  int t = threadIdx.x;
  for (int idx = t; idx < 2048; idx += 512) {
    int b = idx >> 9, j = idx & 511;
    float s = b1[j];
    for (int k = 0; k < 64; ++k) s += lat_sum[b * 64 + k] * w1[k * 512 + j];
    h[idx] = fast_tanh(s);
  }
  __syncthreads();
  if (t < 4) {
    float s = b2[0];
    for (int k = 0; k < 512; ++k) s += h[t * 512 + k] * w2[k];
    out[t] = s;
  }
  for (int i = t; i < 256; i += 512) out[4 + i] = lat_sum[i];
}

// ==========================================================================
extern "C" void kernel_launch(void* const* d_in, const int* in_sizes, int n_in,
                              void* d_out, int out_size, void* d_ws, size_t ws_size,
                              hipStream_t stream) {
  (void)in_sizes; (void)n_in; (void)out_size; (void)ws_size;
  const float* adj       = (const float*)d_in[0];
  const float* node_feat = (const float*)d_in[1];
  const float* edge_feat = (const float*)d_in[2];
  const float* nf1_w = (const float*)d_in[3];  const float* nf1_b = (const float*)d_in[4];
  const float* nf2_w = (const float*)d_in[5];  const float* nf2_b = (const float*)d_in[6];
  const float* ef1_w = (const float*)d_in[7];  const float* ef1_b = (const float*)d_in[8];
  const float* ef2_w = (const float*)d_in[9];  const float* ef2_b = (const float*)d_in[10];
  const float* gcn_w = (const float*)d_in[11];
  const float* vn1_w = (const float*)d_in[12]; const float* vn1_b = (const float*)d_in[13];
  const float* vn1_g = (const float*)d_in[14]; const float* vn1_be = (const float*)d_in[15];
  const float* vn2_w = (const float*)d_in[16]; const float* vn2_b = (const float*)d_in[17];
  const float* vn2_g = (const float*)d_in[18]; const float* vn2_be = (const float*)d_in[19];
  const float* comb_w = (const float*)d_in[20]; const float* comb_b = (const float*)d_in[21];
  const float* lat1_w = (const float*)d_in[22]; const float* lat1_b = (const float*)d_in[23];
  const float* lat2_w = (const float*)d_in[24]; const float* lat2_b = (const float*)d_in[25];
  const float* fc1_w = (const float*)d_in[26]; const float* fc1_b = (const float*)d_in[27];
  const float* fc2_w = (const float*)d_in[28]; const float* fc2_b = (const float*)d_in[29];

  float* ws = (float*)d_ws;
  size_t off = 0;
  float* e_sum  = ws + off; off += 1048576;      // [4,512,512]
  float* alls   = ws + off; off += 2048 * 256;   // [2048,256]
  float* hidden = ws + off; off += 2048 * 64;
  float* hg     = ws + off; off += 2048 * 64;
  float* h1     = ws + off; off += 2048 * 64;
  float* h2     = ws + off; off += 2048 * 64;
  float* gated  = ws + off; off += 2048 * 64;
  float* lath   = ws + off; off += 2048 * 256;
  float* dinv   = ws + off; off += 2048;
  float* vn     = ws + off; off += 256;
  float* lat_sum = ws + off; off += 256;
  unsigned* hgp = (unsigned*)(ws + off); off += 65536;
  unsigned* h2p = (unsigned*)(ws + off); off += 65536;

  k_zero<<<1, 256, 0, stream>>>(lat_sum, 256);
  k_node_mlp<<<16, 256, 0, stream>>>(node_feat, nf1_w, nf1_b, nf2_w, nf2_b,
                                     hidden, alls);
  k_edge_mlp<<<8192, 256, 0, stream>>>(edge_feat, ef1_w, ef1_b, ef2_w, ef2_b,
                                       e_sum);
  k_dinv<<<8, 256, 0, stream>>>(adj, dinv);

  for (int l = 0; l < 3; ++l) {
    k_gemm64<<<16, 256, 0, stream>>>(hidden, gcn_w + l * 64 * 64, hg);
    k_pack<<<256, 256, 0, stream>>>(hg, hgp);
    k_spmm_tanh<<<dim3(4, 4), 256, 0, stream>>>(adj, dinv, hgp, h1);
    k_vn<<<1, 256, 0, stream>>>(h1, vn1_w, vn1_b, vn1_g, vn1_be, vn2_w, vn2_b,
                                vn2_g, vn2_be, vn, l);
    k_addvn<<<256, 256, 0, stream>>>(h1, vn, h2, h2p);
    k_spmm_tanh<<<dim3(4, 4), 256, 0, stream>>>(e_sum, nullptr, h2p, gated);
    k_comb<<<16, 256, 0, stream>>>(h2, gated, comb_w + l * 128 * 64,
                                   comb_b + l * 64, hidden, alls, l);
  }

  k_lat1<<<dim3(16, 4), 256, 0, stream>>>(alls, lat1_w, lat1_b, lath);
  k_lat2<<<16, 256, 0, stream>>>(lath, lat2_w, lat2_b, lat_sum);
  k_fc<<<1, 512, 0, stream>>>(lat_sum, fc1_w, fc1_b, fc2_w, fc2_b,
                              (float*)d_out);
}
